// SG_Attention_18734647345761
// MI455X (gfx1250) — matrix operand order repaired and checked by reading, not compiled, not run on hardware
//
#include <hip/hip_runtime.h>
#include <cstdint>

// ---------------------------------------------------------------------------
// Problem constants: B=8, O=1024, Q=2048, QD=1024, SG=1200 (padded to 1216)
// ---------------------------------------------------------------------------
namespace {
constexpr int kB   = 8;
constexpr int kO   = 1024;
constexpr int kQ   = 2048;
constexpr int kQD  = 1024;
constexpr int kSG  = 1200;
constexpr int kSGP = 1216;   // SG padded to a multiple of KC (zeros)

constexpr int BM   = 128;    // block tile M
constexpr int BN   = 64;     // block tile N
constexpr int KC   = 64;     // K chunk per LDS stage (2 WMMA k-steps)
constexpr int PADU = 8;      // LDS row pad in ushorts (keeps 16B alignment)
constexpr int NT   = 256;    // 8 waves (wave32)
}

#ifndef USE_ASYNC_LDS
#define USE_ASYNC_LDS 1
#endif

typedef __bf16 bf16x16 __attribute__((ext_vector_type(16)));
typedef float  f32x8   __attribute__((ext_vector_type(8)));

union Frag {
  bf16x16 v;
  uint4   u[2];
};

union U4H8 {
  uint4          u;
  unsigned short h[8];
};

__device__ __forceinline__ unsigned short f32_to_bf16(float f) {
  unsigned int x = __float_as_uint(f);
  unsigned int r = x + 0x7fffu + ((x >> 16) & 1u);
  return (unsigned short)(r >> 16);
}

// 16-byte global -> LDS async copy (ASYNCcnt-tracked, no VGPR round trip).
// Generic LDS pointers truncate to the LDS byte offset (ADDR[31:0]).
__device__ __forceinline__ void stage_copy16(unsigned short* lds_dst,
                                             const unsigned short* gsrc) {
#if USE_ASYNC_LDS
  asm volatile("global_load_async_to_lds_b128 %0, %1, off"
               :: "v"((unsigned int)(uintptr_t)lds_dst),
                  "v"((unsigned long long)(uintptr_t)gsrc)
               : "memory");
#else
  *(uint4*)lds_dst = *(const uint4*)gsrc;
#endif
}

__device__ __forceinline__ void stage_fence() {
#if USE_ASYNC_LDS
#if __has_builtin(__builtin_amdgcn_s_wait_asynccnt)
  __builtin_amdgcn_s_wait_asynccnt(0);
#else
  asm volatile("s_wait_asynccnt 0x0" ::: "memory");
#endif
#endif
}

// ---------------------------------------------------------------------------
// f32 -> bf16 conversion passes
// ---------------------------------------------------------------------------
__global__ __launch_bounds__(NT)
void cvt_f32_to_bf16(const float* __restrict__ s, unsigned short* __restrict__ d,
                     long long n4) {
  long long i = (long long)blockIdx.x * NT + threadIdx.x;
  if (i >= n4) return;
  float4 v = *(const float4*)(s + i * 4);
  unsigned short* o = d + i * 4;
  o[0] = f32_to_bf16(v.x);
  o[1] = f32_to_bf16(v.y);
  o[2] = f32_to_bf16(v.z);
  o[3] = f32_to_bf16(v.w);
}

// Convert with zero padding: src [srows][scols] f32 -> dst [drows][dcols] bf16.
__global__ __launch_bounds__(NT)
void cvt_f32_to_bf16_pad(const float* __restrict__ s, unsigned short* __restrict__ d,
                         int srows, int scols, int drows, int dcols) {
  const int c4n = dcols >> 2;
  long long i = (long long)blockIdx.x * NT + threadIdx.x;
  if (i >= (long long)drows * c4n) return;
  const int r = (int)(i / c4n);
  const int c = (int)(i % c4n) << 2;
  float4 v = make_float4(0.f, 0.f, 0.f, 0.f);
  if (r < srows && c < scols)  // scols % 4 == 0
    v = *(const float4*)(s + (long long)r * scols + c);
  unsigned short* o = d + (long long)r * dcols + c;
  o[0] = f32_to_bf16(v.x);
  o[1] = f32_to_bf16(v.y);
  o[2] = f32_to_bf16(v.z);
  o[3] = f32_to_bf16(v.w);
}

// ---------------------------------------------------------------------------
// Stage one K-chunk of A and B into one LDS buffer.
//   B_IS_KxN == false : B[n*ldb + k] -> pure async copy
//   B_IS_KxN == true  : B[k*ldb + n] -> transpose while staging (sync)
// ---------------------------------------------------------------------------
template <bool B_IS_KxN>
__device__ __forceinline__ void stage_tile(
    unsigned short (* __restrict__ as)[KC + PADU],
    unsigned short (* __restrict__ bs)[KC + PADU],
    const unsigned short* __restrict__ A,
    const unsigned short* __restrict__ Bp,
    int mBase, int nBase, int kc, int lda, int ldb, int tid)
{
  // A tile [BM][KC]: 16B async chunks (4 per thread)
#pragma unroll
  for (int t = 0; t < (BM * KC / 8) / NT; ++t) {
    const int idx = tid + t * NT;
    const int r   = idx >> 3;                  // KC/8 == 8 chunks per row
    const int c8  = (idx & 7) << 3;
    stage_copy16(&as[r][c8], A + (long long)(mBase + r) * lda + kc + c8);
  }
  if constexpr (!B_IS_KxN) {
    // B tile [BN][KC]: 16B async chunks (2 per thread)
#pragma unroll
    for (int t = 0; t < (BN * KC / 8) / NT; ++t) {
      const int idx = tid + t * NT;
      const int r   = idx >> 3;
      const int c8  = (idx & 7) << 3;
      stage_copy16(&bs[r][c8], Bp + (long long)(nBase + r) * ldb + kc + c8);
    }
  } else {
    // B stored [K][N]: read 8 bf16 along N, scatter-transpose into LDS
#pragma unroll
    for (int t = 0; t < (KC * (BN / 8)) / NT; ++t) {
      const int idx = tid + t * NT;
      const int kk  = idx >> 3;                // BN/8 == 8 chunks per k-line
      const int n8  = (idx & 7) << 3;
      U4H8 v;
      v.u = *(const uint4*)(Bp + (long long)(kc + kk) * ldb + nBase + n8);
#pragma unroll
      for (int j = 0; j < 8; ++j)
        bs[n8 + j][kk] = v.h[j];
    }
  }
}

// ---------------------------------------------------------------------------
// Double-buffered bf16 WMMA GEMM. A[M,K] bf16 K-contiguous.
// Requires M % BM == 0, K % KC == 0, staged regions fully in-bounds.
//   OUT_BF16: C written as bf16 with zero fill for N <= col < NP, else f32.
// ---------------------------------------------------------------------------
template <bool B_IS_KxN, bool OUT_BF16>
__global__ __launch_bounds__(NT)
void gemm_bf16_wmma(const unsigned short* __restrict__ Ag,
                    const unsigned short* __restrict__ Bg,
                    void* __restrict__ Cg,
                    int M, int N, int NP, int K, int lda, int ldb, int ldc,
                    long long aBatch, long long bBatch, long long cBatch)
{
  __shared__ unsigned short As[2][BM][KC + PADU];   // 2 x 18 KB
  __shared__ unsigned short Bs[2][BN][KC + PADU];   // 2 x  9 KB

  const int tid   = threadIdx.x;
  const int lane  = tid & 31;
  const int wave  = tid >> 5;
  const int waveM = (wave & 3) * 32;   // 4 waves along M
  const int waveN = (wave >> 2) * 32;  // 2 waves along N
  const int lr    = lane & 15;
  const int lh    = lane >> 4;

  const int mBase = blockIdx.y * BM;
  const int nBase = blockIdx.x * BN;

  const unsigned short* A  = Ag + (long long)blockIdx.z * aBatch;
  const unsigned short* Bp = Bg + (long long)blockIdx.z * bBatch;

  const f32x8 zero = {0.f, 0.f, 0.f, 0.f, 0.f, 0.f, 0.f, 0.f};
  f32x8 acc[2][2] = {{zero, zero}, {zero, zero}};

  const int nk = K / KC;

  // Prologue: stage chunk 0 into buffer 0.
  stage_tile<B_IS_KxN>(As[0], Bs[0], A, Bp, mBase, nBase, 0, lda, ldb, tid);

  for (int it = 0; it < nk; ++it) {
    const int cur = it & 1;

    stage_fence();       // my async writes to buf[cur] have landed
    __syncthreads();     // all waves' writes visible; all reads of buf[cur^1]
                         // from the previous iteration are finished

    // Kick off the next chunk's DMA into the other buffer, then compute.
    if (it + 1 < nk)
      stage_tile<B_IS_KxN>(As[cur ^ 1], Bs[cur ^ 1], A, Bp, mBase, nBase,
                           (it + 1) * KC, lda, ldb, tid);

    // ---- 2 WMMA k-steps on buf[cur], overlapped with the async DMA ----
#pragma unroll
    for (int kk = 0; kk < KC; kk += 32) {
      Frag a[2], b[2];
#pragma unroll
      for (int i = 0; i < 2; ++i) {
        const unsigned short* ap = &As[cur][waveM + i * 16 + lr][kk + lh * 8];
        a[i].u[0] = *(const uint4*)(ap);
        a[i].u[1] = *(const uint4*)(ap + 16);
      }
#pragma unroll
      for (int j = 0; j < 2; ++j) {
        const unsigned short* bp = &Bs[cur][waveN + j * 16 + lr][kk + lh * 8];
        b[j].u[0] = *(const uint4*)(bp);
        b[j].u[1] = *(const uint4*)(bp + 16);
      }
#pragma unroll
      for (int i = 0; i < 2; ++i)
#pragma unroll
        for (int j = 0; j < 2; ++j)
          acc[i][j] = __builtin_amdgcn_wmma_f32_16x16x32_bf16(
              false, a[i].v, false, b[j].v, (short)0, acc[i][j], false, false);
    }
  }

  // ---- store C ----
#pragma unroll
  for (int i = 0; i < 2; ++i) {
#pragma unroll
    for (int j = 0; j < 2; ++j) {
      const int col   = nBase + waveN + j * 16 + lr;
      const int rBase = mBase + waveM + i * 16 + lh * 8;
      if constexpr (OUT_BF16) {
        unsigned short* Cb = (unsigned short*)Cg + (long long)blockIdx.z * cBatch;
        if (col < NP) {
#pragma unroll
          for (int v = 0; v < 8; ++v) {
            const unsigned short h =
                (col < N) ? f32_to_bf16(acc[i][j][v]) : (unsigned short)0;
            Cb[(long long)(rBase + v) * ldc + col] = h;
          }
        }
      } else {
        float* Cf = (float*)Cg + (long long)blockIdx.z * cBatch;
        if (col < N) {
#pragma unroll
          for (int v = 0; v < 8; ++v)
            Cf[(long long)(rBase + v) * ldc + col] = acc[i][j][v];
        }
      }
    }
  }
}

// ---------------------------------------------------------------------------
// In-place row softmax over Q=2048; also emits a bf16 copy for the mix GEMM.
// ---------------------------------------------------------------------------
__global__ __launch_bounds__(NT)
void softmax_rows_2048(float* __restrict__ w, unsigned short* __restrict__ wb)
{
  const int row = blockIdx.x;
  float* p = w + (long long)row * kQ;
  unsigned short* pb = wb + (long long)row * kQ;
  const int tid = threadIdx.x;
  constexpr int PER = kQ / NT;  // 8

  float r[PER];
  float vmax = -3.402823466e38f;
#pragma unroll
  for (int i = 0; i < PER; ++i) {
    r[i] = p[tid + i * NT];
    vmax = fmaxf(vmax, r[i]);
  }
#pragma unroll
  for (int off = 16; off > 0; off >>= 1)
    vmax = fmaxf(vmax, __shfl_xor(vmax, off, 32));

  __shared__ float sred[NT / 32];
  const int wv = tid >> 5, ln = tid & 31;
  if (ln == 0) sred[wv] = vmax;
  __syncthreads();
  float m = sred[0];
#pragma unroll
  for (int i = 1; i < NT / 32; ++i) m = fmaxf(m, sred[i]);
  __syncthreads();

  float lsum = 0.f;
#pragma unroll
  for (int i = 0; i < PER; ++i) {
    r[i] = __expf(r[i] - m);
    lsum += r[i];
  }
#pragma unroll
  for (int off = 16; off > 0; off >>= 1)
    lsum += __shfl_xor(lsum, off, 32);
  if (ln == 0) sred[wv] = lsum;
  __syncthreads();
  float s = 0.f;
#pragma unroll
  for (int i = 0; i < NT / 32; ++i) s += sred[i];
  const float inv = 1.0f / s;
#pragma unroll
  for (int i = 0; i < PER; ++i) {
    const float val = r[i] * inv;
    p[tid + i * NT]  = val;
    pb[tid + i * NT] = f32_to_bf16(val);
  }
}

// ---------------------------------------------------------------------------
// d_out = [mix (B,O,SG) | weights (B,O,Q)]
// ---------------------------------------------------------------------------
extern "C" void kernel_launch(void* const* d_in, const int* in_sizes, int n_in,
                              void* d_out, int out_size, void* d_ws, size_t ws_size,
                              hipStream_t stream)
{
  (void)in_sizes; (void)n_in; (void)out_size; (void)ws_size;

  const float* query_in = (const float*)d_in[0];  // [B, O, QD]
  const float* context  = (const float*)d_in[1];  // [B, Q, SG]
  const float* W        = (const float*)d_in[2];  // [SG, QD]

  float* mix     = (float*)d_out;                             // [B, O, SG]
  float* weights = (float*)d_out + (long long)kB * kO * kSG;  // [B, O, Q]

  // Workspace layout (bf16, all 16B-aligned):
  unsigned short* qin_bf = (unsigned short*)d_ws;                  // [B*O][QD]
  unsigned short* w_bf   = qin_bf + (size_t)kB * kO * kQD;         // [SGP][QD] (pad rows zeroed)
  unsigned short* ctx_bf = w_bf   + (size_t)kSGP * kQD;            // [B*Q][SGP] (pad cols zeroed)
  unsigned short* q_bf   = ctx_bf + (size_t)kB * kQ * kSGP;        // [B*O][SGP] (pad cols zeroed)
  unsigned short* wts_bf = q_bf   + (size_t)kB * kO * kSGP;        // [B*O][Q]

  dim3 blk(NT);

  // 0) one-time f32 -> bf16 conversions (bandwidth passes)
  {
    const long long n4 = (long long)kB * kO * kQD / 4;
    cvt_f32_to_bf16<<<dim3((unsigned)((n4 + NT - 1) / NT)), blk, 0, stream>>>(
        query_in, qin_bf, n4);
  }
  {
    const long long tot = (long long)kSGP * (kQD / 4);
    cvt_f32_to_bf16_pad<<<dim3((unsigned)((tot + NT - 1) / NT)), blk, 0, stream>>>(
        W, w_bf, kSG, kQD, kSGP, kQD);
  }
  {
    const long long tot = (long long)kB * kQ * (kSGP / 4);
    cvt_f32_to_bf16_pad<<<dim3((unsigned)((tot + NT - 1) / NT)), blk, 0, stream>>>(
        context, ctx_bf, kB * kQ, kSG, kB * kQ, kSGP);
  }

  // 1) projection: q_bf = qin @ W^T : M=8192, N=1200 (NP=1216), K=1024
  {
    dim3 grid(kSGP / BN, (kB * kO) / BM, 1);
    gemm_bf16_wmma<false, true><<<grid, blk, 0, stream>>>(
        qin_bf, w_bf, q_bf,
        kB * kO, kSG, kSGP, kQD, kQD, kQD, kSGP,
        0LL, 0LL, 0LL);
  }

  // 2) scores: per batch M=O, N=Q, K=SGP (zero-padded, tail-free)
  {
    dim3 grid(kQ / BN, kO / BM, kB);
    gemm_bf16_wmma<false, false><<<grid, blk, 0, stream>>>(
        q_bf, ctx_bf, weights,
        kO, kQ, kQ, kSGP, kSGP, kSGP, kQ,
        (long long)kO * kSGP, (long long)kQ * kSGP, (long long)kO * kQ);
  }

  // 3) softmax (in place) + bf16 copy for the mix GEMM
  softmax_rows_2048<<<dim3(kB * kO), blk, 0, stream>>>(weights, wts_bf);

  // 4) mix: per batch M=O, N=SG, K=Q; B = ctx_bf as [K=Q][N=SGP]
  {
    dim3 grid(kSGP / BN, kO / BM, kB);
    gemm_bf16_wmma<true, false><<<grid, blk, 0, stream>>>(
        wts_bf, ctx_bf, mix,
        kO, kSG, kSGP, kQ, kQ, kSGP, kSG,
        (long long)kO * kQ, (long long)kQ * kSGP, (long long)kO * kSG);
  }
}
